// _VaDER_55370718380163
// MI455X (gfx1250) — matrix operand order repaired and checked by reading, not compiled, not run on hardware
//
#include <hip/hip_runtime.h>
#include <hip/hip_bf16.h>

// ---------------------------------------------------------------------------
// VaDER forward for MI455X (gfx1250), bf16 WMMA path for all large GEMMs.
// ---------------------------------------------------------------------------

namespace {
constexpr int cB = 512, cT = 128, cD = 128, cH = 512, cK = 16, cDM = 64;
constexpr float cEPS = 1e-9f;
constexpr float LOG2PI = 1.8378770664093453f;
}

typedef __attribute__((ext_vector_type(16))) __bf16 v16bf;
typedef __attribute__((ext_vector_type(8)))  float  v8f;
typedef unsigned short u16;

// ---- helpers --------------------------------------------------------------

__device__ __forceinline__ u16 f2bf(float f) {
  unsigned int u = __float_as_uint(f);
  unsigned int r = u + 0x7FFFu + ((u >> 16) & 1u);
  return (u16)(r >> 16);
}

__device__ __forceinline__ float softplusf(float x) {
  return (x > 20.0f) ? x : log1pf(expf(x));
}

__device__ __forceinline__ float sigmf(float x) {
  return 1.0f / (1.0f + expf(-x));
}

// Load 16 bf16 from two 16-byte-aligned chunks into a WMMA fragment register.
__device__ __forceinline__ v16bf ld_frag(const u16* p0, const u16* p1) {
  union { uint4 u[2]; v16bf v; } f;
  f.u[0] = *(const uint4*)p0;
  f.u[1] = *(const uint4*)p1;
  return f.v;
}

#define WMMA_BF16(A, Bf, C) \
  __builtin_amdgcn_wmma_f32_16x16x32_bf16(false, (A), false, (Bf), (short)0, (C), false, false)

// One contiguous K-segment: 2 M-tiles x 4 N-tiles, 8 WMMAs per 32-wide chunk.
// arow0/arow1: per-lane A rows (lane -> M row), wrow: per-lane W row (lane -> N col)
// already offset to this segment's first K column.
__device__ __forceinline__ void mma_span(
    const u16* arow0, const u16* arow1, const u16* wrow, int ldw, int len, int h,
    v8f& c00, v8f& c01, v8f& c02, v8f& c03,
    v8f& c10, v8f& c11, v8f& c12, v8f& c13) {
  for (int kk = 0; kk < len; kk += 32) {
    // A frag: lane holds row m, 16 consecutive K starting at 16*h.
    v16bf a0 = ld_frag(arow0 + kk + 16 * h, arow0 + kk + 16 * h + 8);
    v16bf a1 = ld_frag(arow1 + kk + 16 * h, arow1 + kk + 16 * h + 8);
    // B frags: lane holds column n, 16 consecutive K starting at 16*h.
    const u16* wb = wrow + kk + 16 * h;
    v16bf b0 = ld_frag(wb,                      wb + 8);
    v16bf b1 = ld_frag(wb + (size_t)16 * ldw,   wb + (size_t)16 * ldw + 8);
    v16bf b2 = ld_frag(wb + (size_t)32 * ldw,   wb + (size_t)32 * ldw + 8);
    v16bf b3 = ld_frag(wb + (size_t)48 * ldw,   wb + (size_t)48 * ldw + 8);
    if (kk + 32 < len) __builtin_prefetch(wb + 32, 0, 1);   // global_prefetch
    c00 = WMMA_BF16(a0, b0, c00);
    c10 = WMMA_BF16(a1, b0, c10);
    c01 = WMMA_BF16(a0, b1, c01);
    c11 = WMMA_BF16(a1, b1, c11);
    c02 = WMMA_BF16(a0, b2, c02);
    c12 = WMMA_BF16(a1, b2, c12);
    c03 = WMMA_BF16(a0, b3, c03);
    c13 = WMMA_BF16(a1, b3, c13);
  }
}

// ---- generic WMMA GEMM: out[M,N] = segA[M,Ktot] * W[N,Ktot]^T + bias -------
// A is split into up to 3 row-major bf16 segments along K (for [x|h|c]).
// Block: 256 thr = 8 waves; wave computes a 32x64 output strip (8 wmma accs).

struct GemmArgs {
  const u16* a0; int a0s; int k0;
  const u16* a1; int a1s; int k1;
  const u16* a2; int a2s; int k2;
  const u16* W;  int ldw;
  const float* bias;
  float* out;
  int M, N;
  int mode;   // 0: linear, 1: softplus
};

__device__ __forceinline__ void store_tile(v8f acc, const GemmArgs& g,
                                           int m0, int col0, int h, int l16) {
#pragma unroll
  for (int v = 0; v < 8; ++v) {
    int row = m0 + v + 8 * h;       // C/D layout: lanes 0-15 -> M=v, 16-31 -> M=v+8
    int col = col0 + l16;
    float val = acc[v] + (g.bias ? g.bias[col] : 0.0f);
    if (g.mode == 1) val = softplusf(val);
    g.out[(size_t)row * g.N + col] = val;
  }
}

__global__ __launch_bounds__(256) void k_wmma_gemm(GemmArgs g) {
  const int lane = threadIdx.x & 31;
  const int wave = threadIdx.x >> 5;
  const int h    = lane >> 4;       // lane half selects K sub-block
  const int l16  = lane & 15;
  const int n0   = blockIdx.x * 64;
  const int m0   = blockIdx.y * 256 + wave * 32;   // 2 M-tiles per wave
  if (m0 >= g.M || n0 >= g.N) return;

  v8f c00 = {}, c01 = {}, c02 = {}, c03 = {};
  v8f c10 = {}, c11 = {}, c12 = {}, c13 = {};

  const size_t r0 = (size_t)(m0 + l16);
  const size_t r1 = (size_t)(m0 + 16 + l16);
  const u16* wrow = g.W + (size_t)(n0 + l16) * g.ldw;

  if (g.k0 > 0)
    mma_span(g.a0 + r0 * g.a0s, g.a0 + r1 * g.a0s, wrow, g.ldw, g.k0, h,
             c00, c01, c02, c03, c10, c11, c12, c13);
  if (g.k1 > 0)
    mma_span(g.a1 + r0 * g.a1s, g.a1 + r1 * g.a1s, wrow + g.k0, g.ldw, g.k1, h,
             c00, c01, c02, c03, c10, c11, c12, c13);
  if (g.k2 > 0)
    mma_span(g.a2 + r0 * g.a2s, g.a2 + r1 * g.a2s, wrow + g.k0 + g.k1, g.ldw, g.k2, h,
             c00, c01, c02, c03, c10, c11, c12, c13);

  store_tile(c00, g, m0,      n0 +  0, h, l16);
  store_tile(c01, g, m0,      n0 + 16, h, l16);
  store_tile(c02, g, m0,      n0 + 32, h, l16);
  store_tile(c03, g, m0,      n0 + 48, h, l16);
  store_tile(c10, g, m0 + 16, n0 +  0, h, l16);
  store_tile(c11, g, m0 + 16, n0 + 16, h, l16);
  store_tile(c12, g, m0 + 16, n0 + 32, h, l16);
  store_tile(c13, g, m0 + 16, n0 + 48, h, l16);
}

// ---- decoder recon GEMM fused with masked-MSE accumulation ----------------

struct ReconArgs {
  const u16* A; int as;        // h_bf [B,H]
  const u16* W; int ldw;       // out_W bf16 [D,H]
  const float* bias;           // out_b [D]
  const float* X; const float* Mk; // [B,T,D]
  int t;
  float* sse;
  int M, N, Kt;
};

__device__ __forceinline__ float tile_sse(v8f acc, const ReconArgs& g,
                                          int m0, int col0, int h, int l16) {
  float p = 0.0f;
#pragma unroll
  for (int v = 0; v < 8; ++v) {
    int row = m0 + v + 8 * h;
    int col = col0 + l16;
    float val = acc[v] + g.bias[col];
    size_t xi = ((size_t)row * cT + g.t) * cD + col;
    float e = val - g.X[xi];
    p += g.Mk[xi] * e * e;
  }
  return p;
}

__global__ __launch_bounds__(256) void k_recon(ReconArgs g) {
  const int lane = threadIdx.x & 31;
  const int wave = threadIdx.x >> 5;
  const int h    = lane >> 4;
  const int l16  = lane & 15;
  const int n0   = blockIdx.x * 64;
  const int m0   = blockIdx.y * 128 + wave * 16;
  if (m0 >= g.M || n0 >= g.N) return;

  v8f acc0 = {}, acc1 = {}, acc2 = {}, acc3 = {};
  const u16* arow = g.A + (size_t)(m0 + l16) * g.as;

  for (int kk = 0; kk < g.Kt; kk += 32) {
    v16bf af = ld_frag(arow + kk + 16 * h, arow + kk + 16 * h + 8);
    const u16* wb = g.W + (size_t)(n0 + l16) * g.ldw + kk + 16 * h;
    v16bf b0 = ld_frag(wb,                      wb + 8);
    v16bf b1 = ld_frag(wb + (size_t)16 * g.ldw, wb + (size_t)16 * g.ldw + 8);
    v16bf b2 = ld_frag(wb + (size_t)32 * g.ldw, wb + (size_t)32 * g.ldw + 8);
    v16bf b3 = ld_frag(wb + (size_t)48 * g.ldw, wb + (size_t)48 * g.ldw + 8);
    acc0 = WMMA_BF16(af, b0, acc0);
    acc1 = WMMA_BF16(af, b1, acc1);
    acc2 = WMMA_BF16(af, b2, acc2);
    acc3 = WMMA_BF16(af, b3, acc3);
  }
  float part = tile_sse(acc0, g, m0, n0 +  0, h, l16)
             + tile_sse(acc1, g, m0, n0 + 16, h, l16)
             + tile_sse(acc2, g, m0, n0 + 32, h, l16)
             + tile_sse(acc3, g, m0, n0 + 48, h, l16);
  atomicAdd(g.sse, part);
}

// ---- weight prep ----------------------------------------------------------

// Wcat[4H x (kx+2H)] = [W_ih | W_hh | peephole(W_ch)], bcat = b_ih+b_hh+peep(b_ch)
__global__ __launch_bounds__(256) void k_build_wcat(
    const float* W_ih, const float* W_hh, const float* W_ch,
    const float* b_ih, const float* b_hh, const float* b_ch,
    u16* Wcat, float* bcat, int kx) {
  const int Kt = kx + 2 * cH;
  size_t idx = (size_t)blockIdx.x * 256 + threadIdx.x;
  if (idx >= (size_t)4 * cH * Kt) return;
  int r = (int)(idx / Kt);
  int c = (int)(idx % Kt);
  float v;
  if (c < kx)              v = W_ih[(size_t)r * cD + c];
  else if (c < kx + cH)    v = W_hh[(size_t)r * cH + (c - kx)];
  else {
    int cc = c - kx - cH;
    v = (r < 2 * cH) ? W_ch[(size_t)r * cH + cc]
      : (r < 3 * cH) ? 0.0f
      : W_ch[(size_t)(r - cH) * cH + cc];
  }
  Wcat[idx] = f2bf(v);
  if (c == 0) {
    bcat[r] = b_ih[r] + b_hh[r] +
              ((r < 2 * cH) ? b_ch[r] : (r < 3 * cH) ? 0.0f : b_ch[r - cH]);
  }
}

__global__ __launch_bounds__(256) void k_f2bf(const float* src, u16* dst, int n) {
  int i = blockIdx.x * 256 + threadIdx.x;
  if (i < n) dst[i] = f2bf(src[i]);
}

// ---- imputation: X_imp = X*m + (X @ impute_W^T)*(1-m), stored bf16 --------

__global__ __launch_bounds__(256) void k_impute(const float* X, const float* Mk,
                                                const float* iW, u16* Ximp) {
  size_t idx = (size_t)blockIdx.x * 256 + threadIdx.x;
  if (idx >= (size_t)cB * cT * cD) return;
  int d = (int)(idx % cD);
  size_t r = idx / cD;
  const float* xr = X + r * cD;
  const float* wr = iW + (size_t)d * cD;
  float acc = 0.0f;
#pragma unroll 4
  for (int j = 0; j < cD; ++j) acc += xr[j] * wr[j];
  float xv = X[idx], mv = Mk[idx];
  Ximp[idx] = f2bf(xv * mv + acc * (1.0f - mv));
}

// ---- LSTM cell ------------------------------------------------------------

__global__ __launch_bounds__(256) void k_cell(const float* gbuf, float* c_f,
                                              u16* h_bf, u16* c_bf) {
  int idx = blockIdx.x * 256 + threadIdx.x;
  if (idx >= cB * cH) return;
  int b = idx >> 9;            // /512
  int j = idx & 511;
  const float* gr = gbuf + (size_t)b * 4 * cH;
  float i  = sigmf(gr[j]);
  float f  = sigmf(gr[cH + j]);
  float gg = tanhf(gr[2 * cH + j]);
  float o  = sigmf(gr[3 * cH + j]);
  float cn = f * c_f[idx] + i * gg;
  float hn = o * tanhf(cn);
  c_f[idx] = cn;
  h_bf[idx] = f2bf(hn);
  c_bf[idx] = f2bf(cn);
}

// ---- latent head ----------------------------------------------------------

__global__ __launch_bounds__(256) void k_latent2(
    const float* ef, const float* mu_W, const float* mu_b,
    const float* std_W, const float* std_b, const float* noise,
    float* mu_t, float* std_t, float* var_t, float* z, float* l3_acc) {
  int idx = blockIdx.x * 256 + threadIdx.x;
  if (idx >= cB * cDM) return;
  int d = idx % cDM;
  int b = idx / cDM;
  const float* er = ef + (size_t)b * cH;
  const float* mw = mu_W + (size_t)d * cH;
  const float* sw = std_W + (size_t)d * cH;
  float am = mu_b[d], as = std_b[d];
#pragma unroll 4
  for (int j = 0; j < cH; ++j) { float e = er[j]; am += e * mw[j]; as += e * sw[j]; }
  mu_t[idx] = am;
  std_t[idx] = as;
  var_t[idx] = expf(as);
  z[idx] = am + expf(0.5f * as) * noise[idx];
  atomicAdd(l3_acc, -0.5f * (1.0f + as));
}

__global__ __launch_bounds__(256) void k_latent3(const float* z, const float* W,
                                                 const float* bias, u16* h_bf) {
  int idx = blockIdx.x * 256 + threadIdx.x;
  if (idx >= cB * cH) return;
  int hh = idx & 511;
  int b  = idx >> 9;
  const float* zr = z + (size_t)b * cDM;
  const float* wr = W + (size_t)hh * cDM;
  float a = bias[hh];
#pragma unroll 4
  for (int j = 0; j < cDM; ++j) a += zr[j] * wr[j];
  h_bf[idx] = f2bf(softplusf(a));
}

// ---- GMM loss -------------------------------------------------------------

__global__ void k_gmm_const(const float* var_unsc, const float* phi_unsc,
                            float* var_c, float* lvar_c, float* log_phi) {
  for (int idx = threadIdx.x; idx < cK * cDM; idx += blockDim.x) {
    float v = softplusf(var_unsc[idx]);
    var_c[idx] = v;
    lvar_c[idx] = logf(v + cEPS);
  }
  if (threadIdx.x == 0) {
    float mx = -1e30f;
    for (int k = 0; k < cK; ++k) mx = fmaxf(mx, phi_unsc[k]);
    float s = 0.0f;
    for (int k = 0; k < cK; ++k) s += expf(phi_unsc[k] - mx);
    for (int k = 0; k < cK; ++k) log_phi[k] = logf(expf(phi_unsc[k] - mx) / s + cEPS);
  }
}

__global__ __launch_bounds__(64) void k_gmm_loss(
    const float* z, const float* mu_t, const float* var_t,
    const float* gmm_mu, const float* var_c, const float* lvar_c,
    const float* log_phi, float* l1_acc, float* l2_acc) {
  __shared__ float s1[64], s2[64];
  __shared__ float logp[cK], t123[cK];
  const int b = blockIdx.x, d = threadIdx.x;
  const float zd = z[b * cDM + d];
  const float mt = mu_t[b * cDM + d];
  const float vt = var_t[b * cDM + d];
  for (int k = 0; k < cK; ++k) {
    float mc = gmm_mu[k * cDM + d];
    float vc = var_c[k * cDM + d];
    float lv = lvar_c[k * cDM + d];
    float df = zd - mc;
    float lp = -0.5f * (lv + LOG2PI + df * df / vc);
    float dm = mt - mc;
    float tt = lv + vt / (vc + cEPS) + dm * dm / (vc + cEPS);
    __syncthreads();
    s1[d] = lp; s2[d] = tt;
    __syncthreads();
    for (int off = 32; off > 0; off >>= 1) {
      if (d < off) { s1[d] += s1[d + off]; s2[d] += s2[d + off]; }
      __syncthreads();
    }
    if (d == 0) { logp[k] = log_phi[k] + s1[0]; t123[k] = s2[0]; }
  }
  if (d == 0) {
    float mx = -1e30f;
    for (int k = 0; k < cK; ++k) mx = fmaxf(mx, logp[k]);
    float s = 0.0f;
    for (int k = 0; k < cK; ++k) s += expf(logp[k] - mx);
    float lse = mx + logf(s);
    float l1 = 0.0f, l2 = 0.0f;
    for (int k = 0; k < cK; ++k) {
      float lg = logp[k] - lse;
      float gm = expf(lg);
      l1 += gm * t123[k];
      l2 -= gm * (log_phi[k] - lg);
    }
    atomicAdd(l1_acc, 0.5f * l1);
    atomicAdd(l2_acc, l2);
  }
}

// ---- reductions / final ---------------------------------------------------

__global__ __launch_bounds__(256) void k_mask_sum(const float* m, size_t n, float* acc) {
  __shared__ float sh[256];
  float s = 0.0f;
  for (size_t i = (size_t)blockIdx.x * 256 + threadIdx.x; i < n; i += (size_t)gridDim.x * 256)
    s += m[i];
  sh[threadIdx.x] = s;
  __syncthreads();
  for (int off = 128; off > 0; off >>= 1) {
    if (threadIdx.x < off) sh[threadIdx.x] += sh[threadIdx.x + off];
    __syncthreads();
  }
  if (threadIdx.x == 0) atomicAdd(acc, sh[0]);
}

__global__ void k_final(const float* accs, float* out) {
  // accs: 0=sse 1=mask_sum 2=l1 3=l2 4=l3
  float ms = accs[1];
  float mse = accs[0] / (ms + 1e-12f);
  float rec = mse * (float)(cT * cD) / ms;
  out[0] = rec + (accs[2] + accs[3] + accs[4]) / (float)cB;
}

// ---------------------------------------------------------------------------

extern "C" void kernel_launch(void* const* d_in, const int* in_sizes, int n_in,
                              void* d_out, int out_size, void* d_ws, size_t ws_size,
                              hipStream_t stream) {
  const float* X        = (const float*)d_in[0];
  const float* Mk       = (const float*)d_in[1];
  const float* noise    = (const float*)d_in[2];
  const float* impute_W = (const float*)d_in[3];
  const float* eWih = (const float*)d_in[4],  *eWhh = (const float*)d_in[5],  *eWch = (const float*)d_in[6];
  const float* ebih = (const float*)d_in[7],  *ebhh = (const float*)d_in[8],  *ebch = (const float*)d_in[9];
  const float* dWih = (const float*)d_in[10], *dWhh = (const float*)d_in[11], *dWch = (const float*)d_in[12];
  const float* dbih = (const float*)d_in[13], *dbhh = (const float*)d_in[14], *dbch = (const float*)d_in[15];
  const float* aeW  = (const float*)d_in[16], *aeB  = (const float*)d_in[17];
  const float* adW  = (const float*)d_in[18], *adB  = (const float*)d_in[19];
  const float* muW  = (const float*)d_in[20], *muB  = (const float*)d_in[21];
  const float* stW  = (const float*)d_in[22], *stB  = (const float*)d_in[23];
  const float* outW = (const float*)d_in[24], *outB = (const float*)d_in[25];
  const float* gmmMu = (const float*)d_in[26];
  const float* gmmVar = (const float*)d_in[27];
  const float* gmmPhi = (const float*)d_in[28];

  // workspace carve (256-byte aligned slabs)
  char* ws = (char*)d_ws;
  size_t off = 0;
  auto alloc = [&](size_t bytes) -> void* {
    void* p = ws + off;
    off = (off + bytes + 255) & ~(size_t)255;
    return p;
  };
  const int KtE = cD + 2 * cH;   // 1152
  const int KtD = 2 * cH;        // 1024
  float* accs  = (float*)alloc(8 * sizeof(float));
  u16* Ximp    = (u16*)alloc((size_t)cB * cT * cD * 2);
  u16* WcatE   = (u16*)alloc((size_t)4 * cH * KtE * 2);
  u16* WcatD   = (u16*)alloc((size_t)4 * cH * KtD * 2);
  float* bcatE = (float*)alloc((size_t)4 * cH * 4);
  float* bcatD = (float*)alloc((size_t)4 * cH * 4);
  u16* aeW_bf  = (u16*)alloc((size_t)cH * cH * 2);
  u16* outW_bf = (u16*)alloc((size_t)cD * cH * 2);
  u16* h_bf    = (u16*)alloc((size_t)cB * cH * 2);
  u16* c_bf    = (u16*)alloc((size_t)cB * cH * 2);
  float* c_f   = (float*)alloc((size_t)cB * cH * 4);
  float* gbuf  = (float*)alloc((size_t)cB * 4 * cH * 4);
  float* efeat = (float*)alloc((size_t)cB * cH * 4);
  float* mu_t  = (float*)alloc((size_t)cB * cDM * 4);
  float* std_t = (float*)alloc((size_t)cB * cDM * 4);
  float* var_t = (float*)alloc((size_t)cB * cDM * 4);
  float* zbuf  = (float*)alloc((size_t)cB * cDM * 4);
  float* var_c = (float*)alloc((size_t)cK * cDM * 4);
  float* lvar_c= (float*)alloc((size_t)cK * cDM * 4);
  float* lphi  = (float*)alloc((size_t)cK * 4);
  (void)ws_size; (void)n_in; (void)in_sizes; (void)out_size;

  hipMemsetAsync(accs, 0, 256, stream);

  // weight prep
  {
    size_t nE = (size_t)4 * cH * KtE;
    size_t nD = (size_t)4 * cH * KtD;
    k_build_wcat<<<dim3((unsigned)((nE + 255) / 256)), 256, 0, stream>>>(
        eWih, eWhh, eWch, ebih, ebhh, ebch, WcatE, bcatE, cD);
    k_build_wcat<<<dim3((unsigned)((nD + 255) / 256)), 256, 0, stream>>>(
        dWih, dWhh, dWch, dbih, dbhh, dbch, WcatD, bcatD, 0);
    k_f2bf<<<dim3((cH * cH + 255) / 256), 256, 0, stream>>>(aeW, aeW_bf, cH * cH);
    k_f2bf<<<dim3((cD * cH + 255) / 256), 256, 0, stream>>>(outW, outW_bf, cD * cH);
  }

  // imputation + mask sum
  {
    size_t n = (size_t)cB * cT * cD;
    k_impute<<<dim3((unsigned)((n + 255) / 256)), 256, 0, stream>>>(X, Mk, impute_W, Ximp);
    k_mask_sum<<<dim3(2048), 256, 0, stream>>>(Mk, n, accs + 1);
  }

  // ---- encoder scan ----
  hipMemsetAsync(h_bf, 0, (size_t)cB * cH * 2, stream);
  hipMemsetAsync(c_bf, 0, (size_t)cB * cH * 2, stream);
  hipMemsetAsync(c_f, 0, (size_t)cB * cH * 4, stream);

  GemmArgs ge;
  ge.a0 = Ximp; ge.a0s = cT * cD; ge.k0 = cD;
  ge.a1 = h_bf; ge.a1s = cH;      ge.k1 = cH;
  ge.a2 = c_bf; ge.a2s = cH;      ge.k2 = cH;
  ge.W = WcatE; ge.ldw = KtE; ge.bias = bcatE;
  ge.out = gbuf; ge.M = cB; ge.N = 4 * cH; ge.mode = 0;
  const dim3 gGrid(4 * cH / 64, cB / 256);   // (32,2)
  for (int t = 0; t < cT; ++t) {
    ge.a0 = Ximp + (size_t)t * cD;
    k_wmma_gemm<<<gGrid, 256, 0, stream>>>(ge);
    k_cell<<<dim3(cB * cH / 256), 256, 0, stream>>>(gbuf, c_f, h_bf, c_bf);
  }

  // enc_feat = softplus(c_T @ ae_enc_W^T + b)  (WMMA, A = c_bf)
  {
    GemmArgs gl;
    gl.a0 = c_bf; gl.a0s = cH; gl.k0 = cH;
    gl.a1 = nullptr; gl.a1s = 0; gl.k1 = 0;
    gl.a2 = nullptr; gl.a2s = 0; gl.k2 = 0;
    gl.W = aeW_bf; gl.ldw = cH; gl.bias = aeB;
    gl.out = efeat; gl.M = cB; gl.N = cH; gl.mode = 1;
    k_wmma_gemm<<<dim3(cH / 64, cB / 256), 256, 0, stream>>>(gl);
  }

  // mu/std/z + l3
  k_latent2<<<dim3(cB * cDM / 256), 256, 0, stream>>>(
      efeat, muW, muB, stW, stB, noise, mu_t, std_t, var_t, zbuf, accs + 4);
  // decoder init hidden
  k_latent3<<<dim3(cB * cH / 256), 256, 0, stream>>>(zbuf, adW, adB, h_bf);
  hipMemsetAsync(c_bf, 0, (size_t)cB * cH * 2, stream);
  hipMemsetAsync(c_f, 0, (size_t)cB * cH * 4, stream);

  // GMM constants (independent of decoder)
  k_gmm_const<<<dim3(1), 256, 0, stream>>>(gmmVar, gmmPhi, var_c, lvar_c, lphi);

  // ---- decoder scan (x=0) + fused recon/MSE ----
  GemmArgs gd;
  gd.a0 = h_bf; gd.a0s = cH; gd.k0 = cH;
  gd.a1 = c_bf; gd.a1s = cH; gd.k1 = cH;
  gd.a2 = nullptr; gd.a2s = 0; gd.k2 = 0;
  gd.W = WcatD; gd.ldw = KtD; gd.bias = bcatD;
  gd.out = gbuf; gd.M = cB; gd.N = 4 * cH; gd.mode = 0;

  ReconArgs rr;
  rr.A = h_bf; rr.as = cH;
  rr.W = outW_bf; rr.ldw = cH;
  rr.bias = outB; rr.X = X; rr.Mk = Mk;
  rr.sse = accs + 0; rr.M = cB; rr.N = cD; rr.Kt = cH;
  const dim3 rGrid(cD / 64, cB / 128);       // (2,4)

  for (int t = 0; t < cT; ++t) {
    k_wmma_gemm<<<gGrid, 256, 0, stream>>>(gd);
    k_cell<<<dim3(cB * cH / 256), 256, 0, stream>>>(gbuf, c_f, h_bf, c_bf);
    rr.t = t;
    k_recon<<<rGrid, 256, 0, stream>>>(rr);
  }

  // GMM latent loss
  k_gmm_loss<<<dim3(cB), 64, 0, stream>>>(
      zbuf, mu_t, var_t, gmmMu, var_c, lvar_c, lphi, accs + 2, accs + 3);

  k_final<<<dim3(1), 1, 0, stream>>>(accs, (float*)d_out);
}
